// GPTLanguageModel_21414706937870
// MI455X (gfx1250) — matrix operand order repaired and checked by reading, not compiled, not run on hardware
//
#include <hip/hip_runtime.h>
#include <hip/hip_bf16.h>

// ---------------------------------------------------------------------------
// GPT forward on MI455X (gfx1250, wave32, WMMA + Tensor Data Mover).
// Matmuls: bf16 WMMA (v_wmma_f32_16x16x32_bf16), f32 accumulate.
// Operand tiles: TDM (tensor_load_to_lds) double-buffered into LDS,
// synchronized with s_wait_tensorcnt + block barriers.
// ---------------------------------------------------------------------------

#define B_   2
#define T_   1024
#define D_   1024
#define H_   16
#define KV_  4
#define HS_  64
#define L_   6
#define V_   32000
#define FF_  4096
#define KVD_ (KV_ * HS_)   // 256
#define BT_  (B_ * T_)     // 2048

typedef __attribute__((ext_vector_type(16))) __bf16 v16bf;
typedef __attribute__((ext_vector_type(8)))  float  v8f;
typedef __attribute__((ext_vector_type(4)))  unsigned int u32x4;
typedef __attribute__((ext_vector_type(8)))  int i32x8;
typedef __attribute__((ext_vector_type(4)))  int i32x4;

__device__ __forceinline__ v8f wmma_bf16(v16bf a, v16bf b, v8f c) {
  return __builtin_amdgcn_wmma_f32_16x16x32_bf16(
      /*neg_a=*/false, a, /*neg_b=*/false, b,
      /*c_mod=*/(short)0, c, /*reuse_a=*/false, /*reuse_b=*/false);
}

// A-fragment (16x32, 16-bit) element->K map per CDNA5 ISA 7.12.2.
__device__ __forceinline__ int a_kmap(int e, int half) {
  return e + ((e >= 8) ? 8 : 0) + (half ? 8 : 0);
}

// ---------------------------------------------------------------------------
// TDM 2-D tile load: fp32 tile (tile_w x tile_h, line stride row_stride elems)
// from global into LDS. D# packing per CDNA5 ISA.
// tensor dims == tile dims (tiles are always fully in-bounds here).
// This toolchain's builtin takes 6 args (g0, g1, g2, g3, g4, cpol).
// ---------------------------------------------------------------------------
__device__ __forceinline__ void tdm_load_2d(unsigned lds_off, const float* gptr,
                                            unsigned tile_w, unsigned tile_h,
                                            unsigned long long row_stride) {
  unsigned long long ga = (unsigned long long)(uintptr_t)gptr;
  u32x4 g0;
  g0[0] = 1u;                                            // count=1, user desc
  g0[1] = lds_off;                                       // lds_addr
  g0[2] = (unsigned)ga;                                  // global_addr[31:0]
  g0[3] = (unsigned)((ga >> 32) & 0x1FFFFFFu) | 0x80000000u; // addr[56:32]|type=2
  i32x8 g1;
  g1[0] = (int)(2u << 16);                               // data_size=4B, mask=0
  g1[1] = (int)((tile_w & 0xFFFFu) << 16);               // tensor_dim0 lo16
  g1[2] = (int)((tile_w >> 16) | ((tile_h & 0xFFFFu) << 16)); // dim0 hi|dim1 lo
  g1[3] = (int)((tile_h >> 16) | (tile_w << 16));        // dim1 hi | tile_dim0
  g1[4] = (int)(tile_h & 0xFFFFu);                       // tile_dim1, tile_dim2=0
  g1[5] = (int)(unsigned)(row_stride & 0xFFFFFFFFull);   // dim0_stride lo32
  g1[6] = (int)(unsigned)((row_stride >> 32) & 0xFFFFull); // dim0_stride hi16
  g1[7] = 0;                                             // dim1_stride (unused)
  i32x4 z4 = {0, 0, 0, 0};                               // groups 2/3: 2-D tensor
  i32x8 z8 = {0, 0, 0, 0, 0, 0, 0, 0};
  __builtin_amdgcn_tensor_load_to_lds(g0, g1, z4, z4, z8, 0);
}

__device__ __forceinline__ unsigned lds_off_of(const void* p) {
  return (unsigned)(uintptr_t)p;                         // low 32 bits = LDS offset
}

// ---------------------------------------------------------------------------
// Embedding: x[b,t,:] = tok_emb[tok[b,t],:] + pos_emb[t,:]
// ---------------------------------------------------------------------------
__global__ __launch_bounds__(256) void embed_kernel(
    const int* __restrict__ tok, const float* __restrict__ temb,
    const float* __restrict__ pemb, float* __restrict__ x) {
  int idx = blockIdx.x * 256 + threadIdx.x;
  int d  = idx & (D_ - 1);
  int bt = idx >> 10;
  int t  = bt & (T_ - 1);
  x[idx] = temb[(size_t)tok[bt] * D_ + d] + pemb[(size_t)t * D_ + d];
}

// ---------------------------------------------------------------------------
// LayerNorm over D_=1024, one block (8 waves) per row.
// ---------------------------------------------------------------------------
__global__ __launch_bounds__(256) void ln_kernel(
    const float* __restrict__ x, const float* __restrict__ g,
    const float* __restrict__ b, float* __restrict__ out) {
  int row = blockIdx.x;
  const float* xr = x + (size_t)row * D_;
  float s = 0.f, s2 = 0.f;
  for (int i = threadIdx.x; i < D_; i += 256) {
    float v = xr[i];
    s += v; s2 += v * v;
  }
#pragma unroll
  for (int off = 16; off > 0; off >>= 1) {
    s  += __shfl_xor(s,  off, 32);
    s2 += __shfl_xor(s2, off, 32);
  }
  __shared__ float ls[8], ls2[8];
  int wave = threadIdx.x >> 5;
  if ((threadIdx.x & 31) == 0) { ls[wave] = s; ls2[wave] = s2; }
  __syncthreads();
  float tot = 0.f, tot2 = 0.f;
#pragma unroll
  for (int w = 0; w < 8; ++w) { tot += ls[w]; tot2 += ls2[w]; }
  float mu  = tot * (1.f / D_);
  float var = tot2 * (1.f / D_) - mu * mu;
  float rs  = rsqrtf(var + 1e-5f);
  for (int i = threadIdx.x; i < D_; i += 256)
    out[(size_t)row * D_ + i] = (xr[i] - mu) * rs * g[i] + b[i];
}

// ---------------------------------------------------------------------------
// WMMA GEMM with TDM double-buffered operand staging.
// 256 threads = 8 waves; block tile 128x128; K-step 32.
// Wave grid 2x4: each wave computes 64x32 via 4x2 16x16 accumulators.
// Wave 0 drives the TDM pipeline (EXEC-independent, wave-uniform descriptors).
// ---------------------------------------------------------------------------
template <bool BIAS, bool RELU, bool RES>
__global__ __launch_bounds__(256) void gemm_kernel(
    const float* __restrict__ A, const float* __restrict__ Bm,
    const float* __restrict__ bias, const float* __restrict__ res,
    float* __restrict__ C, int N, int K) {
  __shared__ float Asf[2][128 * 32];   // 32 KB
  __shared__ float Bsf[2][32 * 128];   // 32 KB

  const int bn = blockIdx.x, bm = blockIdx.y;
  const int tid  = threadIdx.x;
  const int wave = tid >> 5;
  const int lane = tid & 31;
  const int half = lane >> 4;
  const int l16  = lane & 15;
  const int wm = (wave >> 2) * 64;
  const int wn = (wave & 3) * 32;

  v8f acc[4][2] = {};
  const int nk = K >> 5;

  if (wave == 0) {  // prologue: stage k-step 0 into buffer 0
    tdm_load_2d(lds_off_of(&Asf[0][0]), A + (size_t)bm * 128 * K, 32, 128, K);
    tdm_load_2d(lds_off_of(&Bsf[0][0]), Bm + bn * 128, 128, 32, N);
  }

  for (int i = 0; i < nk; ++i) {
    const int p = i & 1;
    __syncthreads();   // everyone done reading buffer !p (from step i-1)
    if (wave == 0) {
      if (i + 1 < nk) {
        tdm_load_2d(lds_off_of(&Asf[p ^ 1][0]),
                    A + (size_t)bm * 128 * K + (i + 1) * 32, 32, 128, K);
        tdm_load_2d(lds_off_of(&Bsf[p ^ 1][0]),
                    Bm + (size_t)(i + 1) * 32 * N + bn * 128, 128, 32, N);
        __builtin_amdgcn_s_wait_tensorcnt(2);   // pair for step i complete
      } else {
        __builtin_amdgcn_s_wait_tensorcnt(0);
      }
    }
    __syncthreads();   // buffer p published to all waves

    const float* Ab = Asf[p];
    const float* Bb = Bsf[p];
    v16bf af[4], bf[2];
#pragma unroll
    for (int fi = 0; fi < 4; ++fi) {
      int r = wm + fi * 16 + l16;
#pragma unroll
      for (int e = 0; e < 16; ++e) af[fi][e] = (__bf16)Ab[r * 32 + a_kmap(e, half)];
    }
#pragma unroll
    for (int fj = 0; fj < 2; ++fj) {
      int c = wn + fj * 16 + l16;
#pragma unroll
      for (int e = 0; e < 16; ++e) bf[fj][e] = (__bf16)Bb[a_kmap(e, half) * 128 + c];
    }
#pragma unroll
    for (int fi = 0; fi < 4; ++fi)
#pragma unroll
      for (int fj = 0; fj < 2; ++fj) acc[fi][fj] = wmma_bf16(af[fi], bf[fj], acc[fi][fj]);
  }

  // epilogue: C layout -> lanes 0-15 rows e, lanes 16-31 rows e+8
#pragma unroll
  for (int fi = 0; fi < 4; ++fi)
#pragma unroll
    for (int fj = 0; fj < 2; ++fj) {
      int col = bn * 128 + wn + fj * 16 + l16;
      float bv = BIAS ? bias[col] : 0.f;
#pragma unroll
      for (int e = 0; e < 8; ++e) {
        int row = bm * 128 + wm + fi * 16 + e + 8 * half;
        float v = acc[fi][fj][e] + bv;
        if (RES) v += res[(size_t)row * N + col];
        if (RELU) v = fmaxf(v, 0.f);
        C[(size_t)row * N + col] = v;
      }
    }
}

// ---------------------------------------------------------------------------
// Causal flash attention with GQA. Block = (128-query tile, head, batch);
// 8 waves, 16 query rows each. K/V tiles (32 keys x 64 hs) are TDM-staged
// into double-buffered LDS once per block (trip count is block-uniform).
// S = Q Kt (2 WMMAs / 16-key block), online softmax via 16-lane shfl,
// P re-shaped through wave-private LDS, O += P V (4 WMMAs / 32-key tile).
// ---------------------------------------------------------------------------
__global__ __launch_bounds__(256) void attn_kernel(
    const float* __restrict__ q, const float* __restrict__ kbuf,
    const float* __restrict__ vbuf, float* __restrict__ y) {
  const int qt = blockIdx.x;
  const int h  = blockIdx.y;
  const int b  = blockIdx.z;
  const int kv = h >> 2;                       // GQA: REP = 4
  const int wave = threadIdx.x >> 5;
  const int lane = threadIdx.x & 31;
  const int half = lane >> 4;
  const int l16  = lane & 15;
  const int qbase = qt * 128 + wave * 16;

  __shared__ float  klds[2][32 * 64];          // 16 KB
  __shared__ float  vlds[2][32 * 64];          // 16 KB
  __shared__ __bf16 plds[8][16 * 32];          // 8 KB, wave-private P staging
  __bf16* pw = plds[wave];

  // Q fragments: rows qbase..qbase+15, K-dim = HS in two 32-chunks
  v16bf qa[2];
  {
    const float* qrow = q + (size_t)(b * T_ + qbase + l16) * D_ + h * HS_;
#pragma unroll
    for (int f = 0; f < 2; ++f)
#pragma unroll
      for (int e = 0; e < 16; ++e) qa[f][e] = (__bf16)qrow[a_kmap(e, half) + 32 * f];
  }

  v8f o[4] = {};
  float rmax[8], rsum[8];
#pragma unroll
  for (int e = 0; e < 8; ++e) { rmax[e] = -1e30f; rsum[e] = 0.f; }

  const int ntiles = (qt + 1) * 4;             // block-uniform 32-key tiles
  const float* kbase = kbuf + (size_t)b * T_ * KVD_ + kv * HS_;
  const float* vbase = vbuf + (size_t)b * T_ * KVD_ + kv * HS_;

  if (wave == 0) {                             // prologue: tile 0 into buffer 0
    tdm_load_2d(lds_off_of(&klds[0][0]), kbase, 64, 32, KVD_);
    tdm_load_2d(lds_off_of(&vlds[0][0]), vbase, 64, 32, KVD_);
  }

  for (int jt = 0; jt < ntiles; ++jt) {
    const int k0 = jt * 32;
    const int p = jt & 1;
    __syncthreads();                           // done reading buffer !p
    if (wave == 0) {
      if (jt + 1 < ntiles) {
        tdm_load_2d(lds_off_of(&klds[p ^ 1][0]), kbase + (size_t)(k0 + 32) * KVD_,
                    64, 32, KVD_);
        tdm_load_2d(lds_off_of(&vlds[p ^ 1][0]), vbase + (size_t)(k0 + 32) * KVD_,
                    64, 32, KVD_);
        __builtin_amdgcn_s_wait_tensorcnt(2);
      } else {
        __builtin_amdgcn_s_wait_tensorcnt(0);
      }
    }
    __syncthreads();                           // K/V tile jt published
    const float* Kl = klds[p];
    const float* Vl = vlds[p];

    float sv[2][8];
#pragma unroll
    for (int nb = 0; nb < 2; ++nb) {
      const int key = k0 + nb * 16 + l16;
      v16bf kb0, kb1;
#pragma unroll
      for (int e = 0; e < 16; ++e) {
        int hs = a_kmap(e, half);
        kb0[e] = (__bf16)Kl[(nb * 16 + l16) * 64 + hs];
        kb1[e] = (__bf16)Kl[(nb * 16 + l16) * 64 + hs + 32];
      }
      v8f s = {};
      s = wmma_bf16(qa[0], kb0, s);
      s = wmma_bf16(qa[1], kb1, s);
#pragma unroll
      for (int e = 0; e < 8; ++e) {
        int qr = qbase + e + 8 * half;
        float val = s[e] * 0.125f;             // HS^-0.5
        sv[nb][e] = (key > qr) ? -1e30f : val;
      }
    }
    // online softmax (rows shared across each 16-lane half)
    float scale_r[8];
#pragma unroll
    for (int e = 0; e < 8; ++e) {
      float mx = fmaxf(sv[0][e], sv[1][e]);
#pragma unroll
      for (int off = 1; off < 16; off <<= 1) mx = fmaxf(mx, __shfl_xor(mx, off, 16));
      float mnew = fmaxf(rmax[e], mx);
      float corr = __expf(rmax[e] - mnew);
      float p0 = __expf(sv[0][e] - mnew);
      float p1 = __expf(sv[1][e] - mnew);
      float ps = p0 + p1;
#pragma unroll
      for (int off = 1; off < 16; off <<= 1) ps += __shfl_xor(ps, off, 16);
      rsum[e] = rsum[e] * corr + ps;
      rmax[e] = mnew;
      scale_r[e] = corr;
      sv[0][e] = p0; sv[1][e] = p1;
    }
#pragma unroll
    for (int c = 0; c < 4; ++c)
#pragma unroll
      for (int e = 0; e < 8; ++e) o[c][e] *= scale_r[e];

    // P: C-layout -> LDS -> A-fragment
#pragma unroll
    for (int nb = 0; nb < 2; ++nb)
#pragma unroll
      for (int e = 0; e < 8; ++e)
        pw[(e + 8 * half) * 32 + nb * 16 + l16] = (__bf16)sv[nb][e];
    __syncthreads();
    v16bf pa;
#pragma unroll
    for (int e = 0; e < 16; ++e) pa[e] = pw[l16 * 32 + a_kmap(e, half)];

    // O += P V
#pragma unroll
    for (int c = 0; c < 4; ++c) {
      v16bf vf;
#pragma unroll
      for (int e = 0; e < 16; ++e)
        vf[e] = (__bf16)Vl[a_kmap(e, half) * 64 + c * 16 + l16];
      o[c] = wmma_bf16(pa, vf, o[c]);
    }
  }

#pragma unroll
  for (int c = 0; c < 4; ++c)
#pragma unroll
    for (int e = 0; e < 8; ++e) {
      int m = qbase + e + 8 * half;
      y[(size_t)(b * T_ + m) * D_ + h * HS_ + c * 16 + l16] = o[c][e] / rsum[e];
    }
}

// ---------------------------------------------------------------------------
// Orchestration
// ---------------------------------------------------------------------------
extern "C" void kernel_launch(void* const* d_in, const int* in_sizes, int n_in,
                              void* d_out, int out_size, void* d_ws, size_t ws_size,
                              hipStream_t stream) {
  (void)in_sizes; (void)n_in; (void)out_size; (void)ws_size;
  const int*   tok  = (const int*)d_in[0];
  const float* temb = (const float*)d_in[1];
  const float* pemb = (const float*)d_in[2];
  const float* ln1g = (const float*)d_in[3];
  const float* ln1b = (const float*)d_in[4];
  const float* Wq   = (const float*)d_in[5];
  const float* Wk   = (const float*)d_in[6];
  const float* Wv   = (const float*)d_in[7];
  const float* Wo   = (const float*)d_in[8];
  const float* bo   = (const float*)d_in[9];
  const float* ln2g = (const float*)d_in[10];
  const float* ln2b = (const float*)d_in[11];
  const float* W1   = (const float*)d_in[12];
  const float* b1   = (const float*)d_in[13];
  const float* W2   = (const float*)d_in[14];
  const float* b2   = (const float*)d_in[15];
  const float* lnfg = (const float*)d_in[16];
  const float* lnfb = (const float*)d_in[17];
  const float* Wlm  = (const float*)d_in[18];
  const float* blm  = (const float*)d_in[19];
  float* out = (float*)d_out;

  float* x  = (float*)d_ws;                 // (BT, D) residual stream
  float* hb = x  + (size_t)BT_ * D_;        // (BT, D) ln output
  float* qb = hb + (size_t)BT_ * D_;        // (BT, D) queries
  float* kb = qb + (size_t)BT_ * D_;        // (BT, KVD) keys
  float* vb = kb + (size_t)BT_ * KVD_;      // (BT, KVD) values
  float* yb = vb + (size_t)BT_ * KVD_;      // (BT, D) attention out
  float* ff = yb + (size_t)BT_ * D_;        // (BT, FF) ffn hidden

  const dim3 blk(256);
  const int MB = BT_ / 128;

  embed_kernel<<<BT_ * D_ / 256, blk, 0, stream>>>(tok, temb, pemb, x);

  for (int l = 0; l < L_; ++l) {
    ln_kernel<<<BT_, blk, 0, stream>>>(x, ln1g + l * D_, ln1b + l * D_, hb);
    gemm_kernel<false, false, false><<<dim3(D_ / 128, MB), blk, 0, stream>>>(
        hb, Wq + (size_t)l * D_ * D_, nullptr, nullptr, qb, D_, D_);
    gemm_kernel<false, false, false><<<dim3(KVD_ / 128, MB), blk, 0, stream>>>(
        hb, Wk + (size_t)l * D_ * KVD_, nullptr, nullptr, kb, KVD_, D_);
    gemm_kernel<false, false, false><<<dim3(KVD_ / 128, MB), blk, 0, stream>>>(
        hb, Wv + (size_t)l * D_ * KVD_, nullptr, nullptr, vb, KVD_, D_);
    attn_kernel<<<dim3(T_ / 128, H_, B_), blk, 0, stream>>>(qb, kb, vb, yb);
    gemm_kernel<true, false, true><<<dim3(D_ / 128, MB), blk, 0, stream>>>(
        yb, Wo + (size_t)l * D_ * D_, bo + l * D_, x, x, D_, D_);
    ln_kernel<<<BT_, blk, 0, stream>>>(x, ln2g + l * D_, ln2b + l * D_, hb);
    gemm_kernel<true, true, false><<<dim3(FF_ / 128, MB), blk, 0, stream>>>(
        hb, W1 + (size_t)l * D_ * FF_, b1 + l * FF_, nullptr, ff, FF_, D_);
    gemm_kernel<true, false, true><<<dim3(D_ / 128, MB), blk, 0, stream>>>(
        ff, W2 + (size_t)l * FF_ * D_, b2 + l * D_, x, x, D_, FF_);
  }

  ln_kernel<<<BT_, blk, 0, stream>>>(x, lnfg, lnfb, hb);
  gemm_kernel<true, false, false><<<dim3(V_ / 128, MB), blk, 0, stream>>>(
      hb, Wlm, blm, nullptr, out, V_, D_);
}
